// SelfProjection_31756988186711
// MI455X (gfx1250) — compile-verified, matrix operand order checked by reading, not run on hardware
//
#include <hip/hip_runtime.h>
#include <hip/hip_bf16.h>

// ---------------------------------------------------------------------------
// SelfProjection for MI455X (gfx1250): bf16 WMMA GEMMs + fp32 normalizations.
// B=32, M=N=1024, P=256.
// v4: fix async-to-LDS builtin arg types (int4 AS1 src / int4 AS3 dst),
//     packed ds_store_b64 staging (4x4 register transpose),
//     2x2 register tiling per wave, K-stage 64, global_prefetch.
// ---------------------------------------------------------------------------

#define EPS 1e-5f
#define BB 32
#define MM 1024
#define PP 256

typedef __bf16 v16bf __attribute__((ext_vector_type(16)));
typedef float  v8f   __attribute__((ext_vector_type(8)));
typedef int    v4i   __attribute__((ext_vector_type(4)));

#if defined(__has_builtin)
#if __has_builtin(__builtin_amdgcn_global_load_async_to_lds_b128)
#define HAVE_ASYNC_LDS 1
#endif
#endif

__device__ __forceinline__ __bf16 to_bf16(float f) {
  unsigned u = __builtin_bit_cast(unsigned, f);
  unsigned r = (u + 0x7FFFu + ((u >> 16) & 1u)) >> 16;   // round-to-nearest-even
  return __builtin_bit_cast(__bf16, (unsigned short)r);
}

struct alignas(8) bf4 { __bf16 v[4]; };

#define WMMA_BF16(A, Bv, C) \
  __builtin_amdgcn_wmma_f32_16x16x32_bf16(false, (A), false, (Bv), (short)0, (C), false, false)

// Load a v16bf A/B fragment from an LDS row: elements [k0..k0+7] and [k0+16..k0+23]
__device__ __forceinline__ v16bf frag_ld(const __bf16* p) {
  v16bf f;
#pragma unroll
  for (int i = 0; i < 8; ++i) { f[i] = p[i]; f[8 + i] = p[16 + i]; }
  return f;
}

// ---------------------------------------------------------------------------
// Utility kernels
// ---------------------------------------------------------------------------
__global__ void zero_kernel(float* p, int n) {
  int i = blockIdx.x * 256 + threadIdx.x;
  if (i < n) p[i] = 0.f;
}

// dst[p*1024+m] = bf16(src[m*256+p])   (src: [1024,256] f32, dst: [256,1024] bf16)
__global__ void transpose_bf16_kernel(const float* __restrict__ src,
                                      __bf16* __restrict__ dst) {
  int idx = blockIdx.x * 256 + threadIdx.x;       // 262144 total
  int m = idx >> 8, p = idx & 255;
  dst[(size_t)p * 1024 + m] = to_bf16(src[idx]);
}

// ---------------------------------------------------------------------------
// GEMM1: out[b,r,p] = sum_k A(b,r,k) * W[k,p], A = x or x^T (per batch),
//        1024x1024 x 1024x256. bf16 WMMA, fp32 column-sum via atomics.
// Block: 256 thr = 8 waves (4 row-groups x 2 col-groups);
// tile 128(rows) x 64(cols); K-stage 64; each wave: 32x32 out (4 WMMA accs).
// Grid: (256/64, B*1024/128) = (4, B*8).
// ---------------------------------------------------------------------------
template <bool TRANSA>
__global__ __launch_bounds__(256) void gemm1_kernel(
    const float* __restrict__ x, const float* __restrict__ W,
    __bf16* __restrict__ out, float* __restrict__ colsum) {
  const int pbase = blockIdx.x * 64;
  const int b     = blockIdx.y >> 3;
  const int rbase = (blockIdx.y & 7) * 128;
  const float* xb = x + (size_t)b * MM * MM;

  __shared__ __bf16 As[128][72];   // row stride 144B (16B multiple)
  __shared__ __bf16 Bs[64][72];

  const int tid  = threadIdx.x;
  const int lane = tid & 31, wave = tid >> 5;
  const int wr = (wave & 3) * 32;        // row sub-tile (0,32,64,96)
  const int wc = (wave >> 2) * 32;       // col sub-tile (0,32)
  const int fr = lane & 15;
  const int kh = (lane >> 4) << 3;       // K half: 0 or 8

  v8f acc00 = {}, acc01 = {}, acc10 = {}, acc11 = {};

  for (int kb = 0; kb < MM; kb += 64) {
    // ---- prefetch next K tile (gfx1250 global_prefetch_b8) ----
    if (kb + 64 < MM) {
      if (!TRANSA) {
        int r = tid >> 1;                          // 128 rows, 2 thr/row
        __builtin_prefetch(xb + (size_t)(rbase + r) * MM + kb + 64 + ((tid & 1) << 5), 0, 3);
      } else {
        int k = tid >> 2;                          // 64 rows, 4 thr/row
        __builtin_prefetch(xb + (size_t)(kb + 64 + k) * MM + rbase + ((tid & 3) << 5), 0, 3);
      }
      if (tid < 64) __builtin_prefetch(W + (size_t)(kb + 64 + tid) * PP + pbase, 0, 3);
    }
    // ---- stage A: 128 x 64 ----
    if (!TRANSA) {
      // As[r][k] = x[b][rbase+r][kb+k]; packed 8B LDS stores
#pragma unroll
      for (int it = 0; it < 8; ++it) {
        int idx = it * 256 + tid;          // 0..2047 float4 chunks
        int r  = idx >> 4;                 // 0..127
        int k  = (idx & 15) << 2;          // 0..60
        float4 v = *(const float4*)(xb + (size_t)(rbase + r) * MM + kb + k);
        bf4 t{{to_bf16(v.x), to_bf16(v.y), to_bf16(v.z), to_bf16(v.w)}};
        *(bf4*)&As[r][k] = t;
      }
    } else {
      // As[n][k] = x[b][kb+k][rbase+n]: 4x4 register transpose, 8B LDS stores
      const int n0 = (tid & 31) << 2;      // 0..124
#pragma unroll
      for (int it = 0; it < 2; ++it) {
        int k0 = ((tid >> 5) + it * 8) << 2;   // 0..60
        float4 v0 = *(const float4*)(xb + (size_t)(kb + k0 + 0) * MM + rbase + n0);
        float4 v1 = *(const float4*)(xb + (size_t)(kb + k0 + 1) * MM + rbase + n0);
        float4 v2 = *(const float4*)(xb + (size_t)(kb + k0 + 2) * MM + rbase + n0);
        float4 v3 = *(const float4*)(xb + (size_t)(kb + k0 + 3) * MM + rbase + n0);
        bf4 t0{{to_bf16(v0.x), to_bf16(v1.x), to_bf16(v2.x), to_bf16(v3.x)}};
        bf4 t1{{to_bf16(v0.y), to_bf16(v1.y), to_bf16(v2.y), to_bf16(v3.y)}};
        bf4 t2{{to_bf16(v0.z), to_bf16(v1.z), to_bf16(v2.z), to_bf16(v3.z)}};
        bf4 t3{{to_bf16(v0.w), to_bf16(v1.w), to_bf16(v2.w), to_bf16(v3.w)}};
        *(bf4*)&As[n0 + 0][k0] = t0;
        *(bf4*)&As[n0 + 1][k0] = t1;
        *(bf4*)&As[n0 + 2][k0] = t2;
        *(bf4*)&As[n0 + 3][k0] = t3;
      }
    }
    // ---- stage B transposed: Bs[p][k] = W[kb+k][pbase+p], 4x4 blocks ----
    {
      const int k0 = (tid >> 4) << 2;      // 0..60
      const int p0 = (tid & 15) << 2;      // 0..60
      float4 v0 = *(const float4*)(W + (size_t)(kb + k0 + 0) * PP + pbase + p0);
      float4 v1 = *(const float4*)(W + (size_t)(kb + k0 + 1) * PP + pbase + p0);
      float4 v2 = *(const float4*)(W + (size_t)(kb + k0 + 2) * PP + pbase + p0);
      float4 v3 = *(const float4*)(W + (size_t)(kb + k0 + 3) * PP + pbase + p0);
      bf4 t0{{to_bf16(v0.x), to_bf16(v1.x), to_bf16(v2.x), to_bf16(v3.x)}};
      bf4 t1{{to_bf16(v0.y), to_bf16(v1.y), to_bf16(v2.y), to_bf16(v3.y)}};
      bf4 t2{{to_bf16(v0.z), to_bf16(v1.z), to_bf16(v2.z), to_bf16(v3.z)}};
      bf4 t3{{to_bf16(v0.w), to_bf16(v1.w), to_bf16(v2.w), to_bf16(v3.w)}};
      *(bf4*)&Bs[p0 + 0][k0] = t0;
      *(bf4*)&Bs[p0 + 1][k0] = t1;
      *(bf4*)&Bs[p0 + 2][k0] = t2;
      *(bf4*)&Bs[p0 + 3][k0] = t3;
    }
    __syncthreads();

#pragma unroll
    for (int kk = 0; kk < 64; kk += 32) {
      v16bf a0 = frag_ld(&As[wr      + fr][kk + kh]);
      v16bf a1 = frag_ld(&As[wr + 16 + fr][kk + kh]);
      v16bf b0 = frag_ld(&Bs[wc      + fr][kk + kh]);
      v16bf b1 = frag_ld(&Bs[wc + 16 + fr][kk + kh]);
      acc00 = WMMA_BF16(a0, b0, acc00);
      acc01 = WMMA_BF16(a0, b1, acc01);
      acc10 = WMMA_BF16(a1, b0, acc10);
      acc11 = WMMA_BF16(a1, b1, acc11);
    }
    __syncthreads();
  }

  const int col0 = pbase + wc + fr;
  const int col1 = col0 + 16;
  const int rt   = rbase + wr + ((lane >> 4) << 3);
  float cs0 = 0.f, cs1 = 0.f;
#pragma unroll
  for (int r = 0; r < 8; ++r) {
    float v00 = acc00[r], v01 = acc01[r], v10 = acc10[r], v11 = acc11[r];
    size_t ro0 = (size_t)b * MM * PP + (size_t)(rt + r)      * PP;
    size_t ro1 = (size_t)b * MM * PP + (size_t)(rt + 16 + r) * PP;
    out[ro0 + col0] = to_bf16(v00); out[ro0 + col1] = to_bf16(v01);
    out[ro1 + col0] = to_bf16(v10); out[ro1 + col1] = to_bf16(v11);
    cs0 += v00 + v10; cs1 += v01 + v11;
  }
  atomicAdd(&colsum[b * PP + col0], cs0);
  atomicAdd(&colsum[b * PP + col1], cs1);
}

// ---------------------------------------------------------------------------
// Generic batched GEMM: out[b] = A(b) @ Bm(b), [256 x K] x [K x 256] -> f32.
// A row-major bf16 (row length = K), batch stride aStride (0 = shared).
// A staged via GLOBAL_LOAD_ASYNC_TO_LDS_B128 when available (ASYNCcnt).
// Fused per-batch sum / sum-of-squares. Tile 128x64, K-stage 64, 2x2/wave.
// Grid: (4, 2, B).
// ---------------------------------------------------------------------------
__global__ __launch_bounds__(256) void gemm_bf16_kernel(
    const __bf16* __restrict__ A, size_t aStride,
    const __bf16* __restrict__ Bm, int Ktot,
    float* __restrict__ out, float* __restrict__ stats) {
  const int jbase = blockIdx.x * 64;
  const int ibase = blockIdx.y * 128;
  const int b     = blockIdx.z;
  const __bf16* Ab = A + (size_t)b * aStride;
  const __bf16* Bb = Bm + (size_t)b * (size_t)Ktot * PP;

  __shared__ __bf16 As[128][72];
  __shared__ __bf16 Bs[64][72];

  const int tid  = threadIdx.x;
  const int lane = tid & 31, wave = tid >> 5;
  const int wr = (wave & 3) * 32;
  const int wc = (wave >> 2) * 32;
  const int fr = lane & 15;
  const int kh = (lane >> 4) << 3;

  v8f acc00 = {}, acc01 = {}, acc10 = {}, acc11 = {};

  for (int kb = 0; kb < Ktot; kb += 64) {
    if (kb + 64 < Ktot) {
      int r = tid >> 1;
      __builtin_prefetch(Ab + (size_t)(ibase + r) * Ktot + kb + 64 + ((tid & 1) << 5), 0, 3);
      if (tid < 64) __builtin_prefetch(Bb + (size_t)(kb + 64 + tid) * PP + jbase, 0, 3);
    }
    // ---- stage A: 128 x 64 bf16 (layout-preserving copy) ----
#pragma unroll
    for (int it = 0; it < 4; ++it) {
      int idx = it * 256 + tid;            // 0..1023 chunks of 8 bf16
      int r  = idx >> 3;                   // 0..127
      int kq = (idx & 7) << 3;             // 0..56
      const __bf16* gsrc = Ab + (size_t)(ibase + r) * Ktot + kb + kq;
#if defined(HAVE_ASYNC_LDS)
      __builtin_amdgcn_global_load_async_to_lds_b128(
          (__attribute__((address_space(1))) v4i*)gsrc,
          (__attribute__((address_space(3))) v4i*)&As[r][kq], 0, 0);
#else
      uint4 v = *(const uint4*)gsrc;
      *(uint4*)&As[r][kq] = v;
#endif
    }
    // ---- stage B transposed: Bs[j][k] = Bb[(kb+k)*256 + jbase+j], 4x4 ----
    {
      const int k0 = (tid >> 4) << 2;      // 0..60
      const int j0 = (tid & 15) << 2;      // 0..60
      bf4 r0 = *(const bf4*)(Bb + (size_t)(kb + k0 + 0) * PP + jbase + j0);
      bf4 r1 = *(const bf4*)(Bb + (size_t)(kb + k0 + 1) * PP + jbase + j0);
      bf4 r2 = *(const bf4*)(Bb + (size_t)(kb + k0 + 2) * PP + jbase + j0);
      bf4 r3 = *(const bf4*)(Bb + (size_t)(kb + k0 + 3) * PP + jbase + j0);
      bf4 t0{{r0.v[0], r1.v[0], r2.v[0], r3.v[0]}};
      bf4 t1{{r0.v[1], r1.v[1], r2.v[1], r3.v[1]}};
      bf4 t2{{r0.v[2], r1.v[2], r2.v[2], r3.v[2]}};
      bf4 t3{{r0.v[3], r1.v[3], r2.v[3], r3.v[3]}};
      *(bf4*)&Bs[j0 + 0][k0] = t0;
      *(bf4*)&Bs[j0 + 1][k0] = t1;
      *(bf4*)&Bs[j0 + 2][k0] = t2;
      *(bf4*)&Bs[j0 + 3][k0] = t3;
    }
#if defined(HAVE_ASYNC_LDS)
    asm volatile("s_wait_asynccnt 0" ::: "memory");
#endif
    __syncthreads();

#pragma unroll
    for (int kk = 0; kk < 64; kk += 32) {
      v16bf a0 = frag_ld(&As[wr      + fr][kk + kh]);
      v16bf a1 = frag_ld(&As[wr + 16 + fr][kk + kh]);
      v16bf b0 = frag_ld(&Bs[wc      + fr][kk + kh]);
      v16bf b1 = frag_ld(&Bs[wc + 16 + fr][kk + kh]);
      acc00 = WMMA_BF16(a0, b0, acc00);
      acc01 = WMMA_BF16(a0, b1, acc01);
      acc10 = WMMA_BF16(a1, b0, acc10);
      acc11 = WMMA_BF16(a1, b1, acc11);
    }
    __syncthreads();
  }

  const int col0 = jbase + wc + fr;
  const int col1 = col0 + 16;
  const int rt   = ibase + wr + ((lane >> 4) << 3);
  float s = 0.f, s2 = 0.f;
#pragma unroll
  for (int r = 0; r < 8; ++r) {
    float v00 = acc00[r], v01 = acc01[r], v10 = acc10[r], v11 = acc11[r];
    size_t ro0 = (size_t)b * PP * PP + (size_t)(rt + r)      * PP;
    size_t ro1 = (size_t)b * PP * PP + (size_t)(rt + 16 + r) * PP;
    out[ro0 + col0] = v00; out[ro0 + col1] = v01;
    out[ro1 + col0] = v10; out[ro1 + col1] = v11;
    s  += v00 + v01 + v10 + v11;
    s2 += v00*v00 + v01*v01 + v10*v10 + v11*v11;
  }
  for (int off = 16; off; off >>= 1) {
    s  += __shfl_down(s, off);
    s2 += __shfl_down(s2, off);
  }
  if (lane == 0) {
    atomicAdd(&stats[b * 2 + 0], s);
    atomicAdd(&stats[b * 2 + 1], s2);
  }
}

// ---------------------------------------------------------------------------
// Per-batch layernorm over P=256 (unbiased std), +1. One block per batch.
// ---------------------------------------------------------------------------
__device__ __forceinline__ float block_sum_256(float v, float* red) {
  for (int off = 16; off; off >>= 1) v += __shfl_down(v, off);
  int lane = threadIdx.x & 31, w = threadIdx.x >> 5;
  if (lane == 0) red[w] = v;
  __syncthreads();
  float tot = 0.f;
#pragma unroll
  for (int i = 0; i < 8; ++i) tot += red[i];
  __syncthreads();
  return tot;
}

__global__ __launch_bounds__(256) void layernorm_vec_kernel(
    const float* __restrict__ raw, const float* __restrict__ gamma,
    const float* __restrict__ beta, float* __restrict__ outv) {
  __shared__ float red[8];
  const int b = blockIdx.x, t = threadIdx.x;
  float v = raw[b * PP + t];
  float mean = block_sum_256(v, red) * (1.f / PP);
  float d = v - mean;
  float var = block_sum_256(d * d, red) * (1.f / (PP - 1));
  float istd = 1.f / (sqrtf(var) + EPS);
  outv[b * PP + t] = gamma[t] * (d * istd) + beta[t] + 1.f;
}

// ---------------------------------------------------------------------------
// Elementwise: normalize G_o/G_p (whole-matrix, unbiased), build bf16
// L = Gon*rel, R = Gpn*rel, and write relations to output.
// ---------------------------------------------------------------------------
__global__ __launch_bounds__(256) void build_lr_kernel(
    const float* __restrict__ Go, const float* __restrict__ Gp,
    const float* __restrict__ statsO, const float* __restrict__ statsP,
    const float* __restrict__ osum, const float* __restrict__ psum,
    const float* __restrict__ gamma_o, const float* __restrict__ beta_o,
    const float* __restrict__ gamma_p, const float* __restrict__ beta_p,
    __bf16* __restrict__ L, __bf16* __restrict__ R,
    float* __restrict__ rel_out) {
  size_t idx = (size_t)blockIdx.x * 256 + threadIdx.x;  // B*P*P
  int b = (int)(idx >> 16);
  int ij = (int)(idx & 65535);
  int i = ij >> 8, j = ij & 255;
  const float n = (float)(PP * PP);
  float so1 = statsO[b*2], so2 = statsO[b*2+1];
  float mO = so1 / n;
  float vO = fmaxf((so2 - so1 * so1 / n) / (n - 1.f), 0.f);
  float rO = 1.f / (sqrtf(vO) + EPS);
  float sp1 = statsP[b*2], sp2 = statsP[b*2+1];
  float mP = sp1 / n;
  float vP = fmaxf((sp2 - sp1 * sp1 / n) / (n - 1.f), 0.f);
  float rP = 1.f / (sqrtf(vP) + EPS);

  float gon = gamma_o[ij] * ((Go[idx] - mO) * rO) + beta_o[ij];
  float gpn = gamma_p[ij] * ((Gp[idx] - mP) * rP) + beta_p[ij];
  float rel = osum[b * PP + i] * psum[b * PP + j];
  L[idx] = to_bf16(gon * rel);
  R[idx] = to_bf16(gpn * rel);
  rel_out[idx] = rel;
}

// Final whole-matrix layernorm of projected.
__global__ __launch_bounds__(256) void final_norm_kernel(
    const float* __restrict__ Pm, const float* __restrict__ stats,
    const float* __restrict__ gamma, const float* __restrict__ beta,
    float* __restrict__ out) {
  size_t idx = (size_t)blockIdx.x * 256 + threadIdx.x;
  int b = (int)(idx >> 16);
  int ij = (int)(idx & 65535);
  const float n = (float)(PP * PP);
  float s1 = stats[b*2], s2 = stats[b*2+1];
  float m = s1 / n;
  float v = fmaxf((s2 - s1 * s1 / n) / (n - 1.f), 0.f);
  float r = 1.f / (sqrtf(v) + EPS);
  out[idx] = gamma[ij] * ((Pm[idx] - m) * r) + beta[ij];
}

// ---------------------------------------------------------------------------
// Host launcher
// ---------------------------------------------------------------------------
extern "C" void kernel_launch(void* const* d_in, const int* in_sizes, int n_in,
                              void* d_out, int out_size, void* d_ws, size_t ws_size,
                              hipStream_t stream) {
  const float* x    = (const float*)d_in[0];
  const float* Wo   = (const float*)d_in[1];   // original_xj_y [1024,256]
  const float* Xi   = (const float*)d_in[2];   // original_xi_y [1024,256]
  const float* Wp   = (const float*)d_in[3];   // permuted_xj_y [1024,256]
  const float* Xpi  = (const float*)d_in[4];   // permuted_xi_y [1024,256]
  const float* g_os = (const float*)d_in[5];
  const float* g_o  = (const float*)d_in[6];
  const float* g_ps = (const float*)d_in[7];
  const float* g_p  = (const float*)d_in[8];
  const float* g_f  = (const float*)d_in[9];
  const float* b_os = (const float*)d_in[10];
  const float* b_o  = (const float*)d_in[11];
  const float* b_ps = (const float*)d_in[12];
  const float* b_p  = (const float*)d_in[13];
  const float* b_f  = (const float*)d_in[14];
  (void)in_sizes; (void)n_in; (void)out_size; (void)ws_size;

  char* ws = (char*)d_ws;
  size_t off = 0;
  auto take = [&](size_t bytes) -> char* {
    char* p = ws + off;
    off = (off + bytes + 255) & ~(size_t)255;
    return p;
  };
  const size_t nBMP = (size_t)BB * MM * PP;    // 8,388,608
  const size_t nBPP = (size_t)BB * PP * PP;    // 2,097,152

  __bf16* oyy   = (__bf16*)take(nBMP * 2);
  __bf16* pyy   = (__bf16*)take(nBMP * 2);
  __bf16* XiT   = (__bf16*)take((size_t)PP * MM * 2);
  __bf16* XpiT  = (__bf16*)take((size_t)PP * MM * 2);
  float*  osraw = (float*)take((size_t)BB * PP * 4);
  float*  psraw = (float*)take((size_t)BB * PP * 4);
  float*  osum  = (float*)take((size_t)BB * PP * 4);
  float*  psum  = (float*)take((size_t)BB * PP * 4);
  float*  Go    = (float*)take(nBPP * 4);
  float*  Gp    = (float*)take(nBPP * 4);
  float*  stats = (float*)take((size_t)6 * BB * 4);  // [O:2B][P:2B][proj:2B]
  __bf16* Lm    = (__bf16*)take(nBPP * 2);
  __bf16* Rm    = (__bf16*)take(nBPP * 2);
  float*  Pm    = (float*)take(nBPP * 4);

  float* statsO  = stats;
  float* statsP  = stats + 2 * BB;
  float* statsPr = stats + 4 * BB;

  float* out_proj = (float*)d_out;             // [B,P,P]
  float* out_rel  = out_proj + nBPP;           // [B,P,P]

  // 0) zero accumulators
  zero_kernel<<<dim3((BB * PP + 255) / 256), dim3(256), 0, stream>>>(osraw, BB * PP);
  zero_kernel<<<dim3((BB * PP + 255) / 256), dim3(256), 0, stream>>>(psraw, BB * PP);
  zero_kernel<<<dim3(1), dim3(256), 0, stream>>>(stats, 6 * BB);

  // 1) transpose-convert Xi, Xpi -> bf16 [256,1024]
  transpose_bf16_kernel<<<dim3(1024), dim3(256), 0, stream>>>(Xi, XiT);
  transpose_bf16_kernel<<<dim3(1024), dim3(256), 0, stream>>>(Xpi, XpiT);

  // 2) big GEMMs over x (original + permuted paths), fused column-sums
  gemm1_kernel<false><<<dim3(PP / 64, BB * 8), dim3(256), 0, stream>>>(x, Wo, oyy, osraw);
  gemm1_kernel<true ><<<dim3(PP / 64, BB * 8), dim3(256), 0, stream>>>(x, Wp, pyy, psraw);

  // 3) layernorm of the column sums (+1)
  layernorm_vec_kernel<<<dim3(BB), dim3(256), 0, stream>>>(osraw, g_os, b_os, osum);
  layernorm_vec_kernel<<<dim3(BB), dim3(256), 0, stream>>>(psraw, g_ps, b_ps, psum);

  // 4) G_o = Xi^T @ oyy, G_p = Xpi^T @ pyy  (+ per-batch sum/sumsq)
  gemm_bf16_kernel<<<dim3(PP / 64, 2, BB), dim3(256), 0, stream>>>(
      XiT, 0, oyy, MM, Go, statsO);
  gemm_bf16_kernel<<<dim3(PP / 64, 2, BB), dim3(256), 0, stream>>>(
      XpiT, 0, pyy, MM, Gp, statsP);

  // 5) normalize + relations + bf16 operands for final GEMM
  build_lr_kernel<<<dim3((unsigned)(nBPP / 256)), dim3(256), 0, stream>>>(
      Go, Gp, statsO, statsP, osum, psum, g_o, b_o, g_p, b_p, Lm, Rm, out_rel);

  // 6) projected = L @ R  (+ stats)
  gemm_bf16_kernel<<<dim3(PP / 64, 2, BB), dim3(256), 0, stream>>>(
      Lm, (size_t)PP * PP, Rm, PP, Pm, statsPr);

  // 7) final whole-matrix layernorm -> output
  final_norm_kernel<<<dim3((unsigned)(nBPP / 256)), dim3(256), 0, stream>>>(
      Pm, statsPr, g_f, b_f, out_proj);
}
